// ContextualNodeModel_49976239456339
// MI455X (gfx1250) — matrix operand order repaired and checked by reading, not compiled, not run on hardware
//
#include <hip/hip_runtime.h>
#include <hip/hip_bf16.h>

typedef __bf16 bf16_t;
typedef __attribute__((ext_vector_type(8)))  bf16_t v8bf;
typedef __attribute__((ext_vector_type(16))) bf16_t v16bf;
typedef __attribute__((ext_vector_type(8)))  float  v8f;

#define NNODES   50000
#define NEDGES   200000
#define NFEDGES  100000
#define DNODE    256
#define DEDGE    32
#define DFIN     544   // 2*256 + 32
#define DFHID    256
#define DFOUT    128
#define DTIN     384
#define DTHID    512
#define DTOUT    256

// ---------------- WMMA fragment helpers (wave32, 16x16x32 bf16) ----------------

static __device__ __forceinline__ v16bf cat16(v8bf lo, v8bf hi) {
  return __builtin_shufflevector(lo, hi, 0,1,2,3,4,5,6,7,8,9,10,11,12,13,14,15);
}

// A: 16x32 bf16 tile, row-major source with row stride `stride` (elements).
// lane<16: K {k0..k0+7, k0+16..k0+23}; lane>=16: +8.  Two 16B loads.
static __device__ __forceinline__ v16bf load_a_frag(const bf16_t* base, int stride,
                                                    int m_tile, int k0, int lane) {
  int m  = m_tile * 16 + (lane & 15);
  int hi = lane >> 4;
  const bf16_t* p = base + (size_t)m * stride + k0 + hi * 8;
  v8bf lo = *(const v8bf*)p;
  v8bf hh = *(const v8bf*)(p + 16);
  return cat16(lo, hh);
}

// B: 32x16 bf16 tile from transposed weights Wt[n][k] (row stride K).
// lane n = l&15 holds K {k0..k0+7, k0+16..k0+23} (+8 for lanes>=16).
static __device__ __forceinline__ v16bf load_b_frag(const bf16_t* Wt, int K,
                                                    int n_tile, int k0, int lane) {
  int n  = n_tile * 16 + (lane & 15);
  int kb = k0 + ((lane >> 4) << 3);
  const bf16_t* p = Wt + (size_t)n * K + kb;
  v8bf lo = *(const v8bf*)p;
  v8bf hh = *(const v8bf*)(p + 16);
  return cat16(lo, hh);
}

static __device__ __forceinline__ v8f wmma_bf16(v16bf a, v16bf b, v8f c) {
  return __builtin_amdgcn_wmma_f32_16x16x32_bf16(false, a, false, b, (short)0, c,
                                                 false, false);
}

static __device__ __forceinline__ v8f vzero8() {
  v8f z = {0.f,0.f,0.f,0.f,0.f,0.f,0.f,0.f};
  return z;
}

// ---------------- conversion / init kernels ----------------

__global__ void cvt_x_kernel(const float* __restrict__ in, bf16_t* __restrict__ out, int n4) {
  int i = blockIdx.x * blockDim.x + threadIdx.x;
  int stride = gridDim.x * blockDim.x;
  for (; i < n4; i += stride) {
    float4 f = ((const float4*)in)[i];
    bf16_t* d = out + (size_t)i * 4;
    d[0] = (bf16_t)f.x; d[1] = (bf16_t)f.y; d[2] = (bf16_t)f.z; d[3] = (bf16_t)f.w;
  }
}

// in: f32 [K][N] -> out: bf16 [N][K] (transposed).  swap!=0 swaps the first two
// 256-row input blocks (used to make Wb1 consume [x_future,x_past] ordering).
__global__ void cvt_wt_kernel(const float* __restrict__ in, bf16_t* __restrict__ out,
                              int K, int N, int swap) {
  int i = blockIdx.x * blockDim.x + threadIdx.x;
  int stride = gridDim.x * blockDim.x;
  int total = K * N;
  for (; i < total; i += stride) {
    int n = i / K;
    int k = i - n * K;
    int ks = k;
    if (swap) ks = (k < 256) ? k + 256 : ((k < 512) ? k - 256 : k);
    out[(size_t)n * K + k] = (bf16_t)in[(size_t)ks * N + n];
  }
}

// ---------------- fused edge MLP + scatter-add ----------------
// 64 edges / block, 256 threads (8 waves).  Stages concat([xA, xB, attr]) as
// bf16 in LDS, runs 544->256 (relu) ->128 via WMMA, scatter-adds to agg.
// DUAL=true : two weight sets sharing the staged input (fwd + swapped-bwd),
//             model0 scatters via idxA/colA, model1 via idxB/colB.
// DUAL=false: one model, output scattered to BOTH idxA/colA and idxB/colB.

template<bool DUAL>
__global__ __launch_bounds__(256)
void edge_mlp_kernel(const bf16_t* __restrict__ xbf,
                     const int* __restrict__ idxA, const int* __restrict__ idxB,
                     const float* __restrict__ attr,
                     const bf16_t* __restrict__ W1A, const bf16_t* __restrict__ W1B,
                     const float* __restrict__ b1A, const float* __restrict__ b1B,
                     const bf16_t* __restrict__ W2A, const bf16_t* __restrict__ W2B,
                     const float* __restrict__ b2A, const float* __restrict__ b2B,
                     float* __restrict__ agg, int nE, int colA, int colB)
{
  constexpr int SA = 560;   // staged input row stride (>=544), 16B-aligned rows
  constexpr int SH = 264;   // hidden row stride (>=256), 16B-aligned rows
  __shared__ __align__(16) bf16_t smem[64 * SA];   // 71.7 KB; hidden overlays it
  __shared__ int la[64], lb[64];

  const int tid = threadIdx.x;
  const int e0  = blockIdx.x * 64;

  if (tid < 64) {
    int e = e0 + tid;
    la[tid] = (e < nE) ? idxA[e] : 0;
    lb[tid] = (e < nE) ? idxB[e] : 0;
  }
  __syncthreads();

  // ---- stage gathered node rows (already bf16) : 64 edges x 2 rows x 256 ----
  #pragma unroll
  for (int it = 0; it < 16; ++it) {
    int c      = tid + it * 256;       // 16B-chunk id, 0..4095
    int e_loc  = c >> 6;
    int within = c & 63;
    int part   = within >> 5;          // 0: xA cols 0..255, 1: xB cols 256..511
    int col8   = (within & 31) << 3;
    int row    = part ? lb[e_loc] : la[e_loc];
    uint4 v = *(const uint4*)(xbf + (size_t)row * DNODE + col8);
    *(uint4*)(smem + (size_t)e_loc * SA + part * DNODE + col8) = v;
  }
  // ---- stage edge attr (f32 -> bf16), cols 512..543 ----
  {
    int idx   = tid << 3;              // 8 floats per thread = 64*32 total
    int e_loc = idx >> 5;
    int cs    = idx & 31;
    int e     = e0 + e_loc;
    float f[8] = {0.f,0.f,0.f,0.f,0.f,0.f,0.f,0.f};
    if (e < nE) {
      const float4* s = (const float4*)(attr + (size_t)e * DEDGE + cs);
      float4 a = s[0], b = s[1];
      f[0]=a.x; f[1]=a.y; f[2]=a.z; f[3]=a.w; f[4]=b.x; f[5]=b.y; f[6]=b.z; f[7]=b.w;
    }
    bf16_t* d = smem + (size_t)e_loc * SA + 512 + cs;
    #pragma unroll
    for (int i = 0; i < 8; ++i) d[i] = (bf16_t)f[i];
  }
  __syncthreads();

  const int w      = tid >> 5;   // wave id (wave32)
  const int l      = tid & 31;
  const int m_tile = w >> 1;     // 4 x 16-edge M tiles
  const int nhalf  = w & 1;      // which 8 of 16 hidden N tiles

  // ---- layer 1: [64 x 544] @ [544 x 256] ----
  v8f cf[8], cb[8];
  #pragma unroll
  for (int i = 0; i < 8; ++i) { cf[i] = vzero8(); cb[i] = vzero8(); }
  for (int ks = 0; ks < DFIN / 32; ++ks) {       // 17 K-steps
    int k0 = ks * 32;
    v16bf a = load_a_frag(smem, SA, m_tile, k0, l);
    #pragma unroll
    for (int nt = 0; nt < 8; ++nt) {
      int n_tile = nhalf * 8 + nt;
      v16bf bA = load_b_frag(W1A, DFIN, n_tile, k0, l);
      cf[nt] = wmma_bf16(a, bA, cf[nt]);
      if constexpr (DUAL) {
        v16bf bB = load_b_frag(W1B, DFIN, n_tile, k0, l);
        cb[nt] = wmma_bf16(a, bB, cb[nt]);
      }
    }
  }
  __syncthreads();   // staged input no longer needed; overlay hidden

  bf16_t* Hf = smem;
  bf16_t* Hb = smem + 64 * SH;
  {
    int hi = l >> 4, nn = l & 15;
    #pragma unroll
    for (int nt = 0; nt < 8; ++nt) {
      int n = (nhalf * 8 + nt) * 16 + nn;
      float bvA = b1A[n];
      float bvB = DUAL ? b1B[n] : 0.f;
      #pragma unroll
      for (int v = 0; v < 8; ++v) {
        int r = m_tile * 16 + hi * 8 + v;
        float xf = cf[nt][v] + bvA; xf = xf > 0.f ? xf : 0.f;
        Hf[(size_t)r * SH + n] = (bf16_t)xf;
        if constexpr (DUAL) {
          float xb = cb[nt][v] + bvB; xb = xb > 0.f ? xb : 0.f;
          Hb[(size_t)r * SH + n] = (bf16_t)xb;
        }
      }
    }
  }
  __syncthreads();

  // ---- layer 2: [64 x 256] @ [256 x 128] + scatter ----
  if constexpr (DUAL) {
    const int model      = w & 1;
    const bf16_t* Hm     = model ? Hb  : Hf;
    const bf16_t* W2     = model ? W2B : W2A;
    const float*  b2     = model ? b2B : b2A;
    const int*    sidx   = model ? lb  : la;
    const int     colbas = model ? colB : colA;
    v8f o[8];
    #pragma unroll
    for (int i = 0; i < 8; ++i) o[i] = vzero8();
    for (int ks = 0; ks < DFHID / 32; ++ks) {
      int k0 = ks * 32;
      v16bf a = load_a_frag(Hm, SH, m_tile, k0, l);
      #pragma unroll
      for (int nt = 0; nt < 8; ++nt) {
        v16bf b = load_b_frag(W2, DFHID, nt, k0, l);
        o[nt] = wmma_bf16(a, b, o[nt]);
      }
    }
    int hi = l >> 4, nn = l & 15;
    #pragma unroll
    for (int nt = 0; nt < 8; ++nt) {
      int n = nt * 16 + nn;
      float bv = b2[n];
      #pragma unroll
      for (int v = 0; v < 8; ++v) {
        int e_loc = m_tile * 16 + hi * 8 + v;
        if (e0 + e_loc < nE)
          unsafeAtomicAdd(&agg[(size_t)sidx[e_loc] * DTIN + colbas + n], o[nt][v] + bv);
      }
    }
  } else {
    v8f o[4];
    #pragma unroll
    for (int i = 0; i < 4; ++i) o[i] = vzero8();
    for (int ks = 0; ks < DFHID / 32; ++ks) {
      int k0 = ks * 32;
      v16bf a = load_a_frag(Hf, SH, m_tile, k0, l);
      #pragma unroll
      for (int nt = 0; nt < 4; ++nt) {
        v16bf b = load_b_frag(W2A, DFHID, nhalf * 4 + nt, k0, l);
        o[nt] = wmma_bf16(a, b, o[nt]);
      }
    }
    int hi = l >> 4, nn = l & 15;
    #pragma unroll
    for (int nt = 0; nt < 4; ++nt) {
      int n = (nhalf * 4 + nt) * 16 + nn;
      float bv = b2A[n];
      #pragma unroll
      for (int v = 0; v < 8; ++v) {
        int e_loc = m_tile * 16 + hi * 8 + v;
        if (e0 + e_loc < nE) {
          float val = o[nt][v] + bv;
          unsafeAtomicAdd(&agg[(size_t)la[e_loc] * DTIN + colA + n], val);
          unsafeAtomicAdd(&agg[(size_t)lb[e_loc] * DTIN + colB + n], val);
        }
      }
    }
  }
}

// ---------------- final node MLP: [N x 384] -> 512 (relu) -> 256 ----------------

__global__ __launch_bounds__(256)
void total_mlp_kernel(const float* __restrict__ agg,
                      const bf16_t* __restrict__ W1t, const float* __restrict__ b1,
                      const bf16_t* __restrict__ W2t, const float* __restrict__ b2,
                      float* __restrict__ out, int N)
{
  constexpr int SA = 400;   // staged input stride (>=384)
  constexpr int SH = 528;   // hidden stride (>=512)
  __shared__ __align__(16) bf16_t smem[64 * SH];   // 67.6 KB; input overlays it

  const int tid = threadIdx.x;
  const int n0  = blockIdx.x * 64;

  // stage agg rows (f32 -> bf16): 64 x 384 = 3072 chunks of 8
  #pragma unroll
  for (int it = 0; it < 12; ++it) {
    int c    = tid + it * 256;
    int r    = c / 48;
    int col8 = (c % 48) << 3;
    int node = n0 + r;
    float f[8] = {0.f,0.f,0.f,0.f,0.f,0.f,0.f,0.f};
    if (node < N) {
      const float4* s = (const float4*)(agg + (size_t)node * DTIN + col8);
      float4 a = s[0], b = s[1];
      f[0]=a.x; f[1]=a.y; f[2]=a.z; f[3]=a.w; f[4]=b.x; f[5]=b.y; f[6]=b.z; f[7]=b.w;
    }
    bf16_t* d = smem + (size_t)r * SA + col8;
    #pragma unroll
    for (int i = 0; i < 8; ++i) d[i] = (bf16_t)f[i];
  }
  __syncthreads();

  const int w = tid >> 5, l = tid & 31;
  const int m_tile = w >> 1, nhalf = w & 1;

  // layer 1: [64 x 384] @ [384 x 512]
  v8f h[16];
  #pragma unroll
  for (int i = 0; i < 16; ++i) h[i] = vzero8();
  for (int ks = 0; ks < DTIN / 32; ++ks) {       // 12 K-steps
    int k0 = ks * 32;
    v16bf a = load_a_frag(smem, SA, m_tile, k0, l);
    #pragma unroll
    for (int nt = 0; nt < 16; ++nt) {
      v16bf b = load_b_frag(W1t, DTIN, nhalf * 16 + nt, k0, l);
      h[nt] = wmma_bf16(a, b, h[nt]);
    }
  }
  __syncthreads();
  {
    int hi = l >> 4, nn = l & 15;
    #pragma unroll
    for (int nt = 0; nt < 16; ++nt) {
      int n = (nhalf * 16 + nt) * 16 + nn;
      float bv = b1[n];
      #pragma unroll
      for (int v = 0; v < 8; ++v) {
        int r = m_tile * 16 + hi * 8 + v;
        float xv = h[nt][v] + bv; xv = xv > 0.f ? xv : 0.f;
        smem[(size_t)r * SH + n] = (bf16_t)xv;
      }
    }
  }
  __syncthreads();

  // layer 2: [64 x 512] @ [512 x 256]
  v8f o[8];
  #pragma unroll
  for (int i = 0; i < 8; ++i) o[i] = vzero8();
  for (int ks = 0; ks < DTHID / 32; ++ks) {      // 16 K-steps
    int k0 = ks * 32;
    v16bf a = load_a_frag(smem, SH, m_tile, k0, l);
    #pragma unroll
    for (int nt = 0; nt < 8; ++nt) {
      v16bf b = load_b_frag(W2t, DTHID, nhalf * 8 + nt, k0, l);
      o[nt] = wmma_bf16(a, b, o[nt]);
    }
  }
  {
    int hi = l >> 4, nn = l & 15;
    #pragma unroll
    for (int nt = 0; nt < 8; ++nt) {
      int n = (nhalf * 8 + nt) * 16 + nn;
      float bv = b2[n];
      #pragma unroll
      for (int v = 0; v < 8; ++v) {
        int node = n0 + m_tile * 16 + hi * 8 + v;
        if (node < N) out[(size_t)node * DTOUT + n] = o[nt][v] + bv;
      }
    }
  }
}

// ---------------- launcher ----------------

extern "C" void kernel_launch(void* const* d_in, const int* in_sizes, int n_in,
                              void* d_out, int out_size, void* d_ws, size_t ws_size,
                              hipStream_t stream) {
  (void)in_sizes; (void)n_in; (void)out_size; (void)ws_size;

  const float* x     = (const float*)d_in[0];
  const int*   ei    = (const int*)  d_in[1];   // [0]=past (row0), [1]=future (row1)
  const float* eatt  = (const float*)d_in[2];
  const int*   fei   = (const int*)  d_in[3];   // [0]=early, [1]=later
  const float* fatt  = (const float*)d_in[4];
  const float* Wf1 = (const float*)d_in[5];  const float* bf1 = (const float*)d_in[6];
  const float* Wf2 = (const float*)d_in[7];  const float* bf2 = (const float*)d_in[8];
  const float* Wb1 = (const float*)d_in[9];  const float* bb1 = (const float*)d_in[10];
  const float* Wb2 = (const float*)d_in[11]; const float* bb2 = (const float*)d_in[12];
  const float* Wr1 = (const float*)d_in[13]; const float* br1 = (const float*)d_in[14];
  const float* Wr2 = (const float*)d_in[15]; const float* br2 = (const float*)d_in[16];
  const float* Wt1 = (const float*)d_in[17]; const float* bt1 = (const float*)d_in[18];
  const float* Wt2 = (const float*)d_in[19]; const float* bt2 = (const float*)d_in[20];

  // workspace carve (all offsets 256B-aligned)
  char* ws = (char*)d_ws;
  size_t off = 0;
  float*  agg  = (float*)(ws + off);  off += (size_t)NNODES * DTIN * 4;   // 76.8 MB
  bf16_t* xbf  = (bf16_t*)(ws + off); off += (size_t)NNODES * DNODE * 2;  // 25.6 MB
  bf16_t* Wf1t = (bf16_t*)(ws + off); off += (size_t)DFHID * DFIN * 2;
  bf16_t* Wb1t = (bf16_t*)(ws + off); off += (size_t)DFHID * DFIN * 2;
  bf16_t* Wr1t = (bf16_t*)(ws + off); off += (size_t)DFHID * DFIN * 2;
  bf16_t* Wf2t = (bf16_t*)(ws + off); off += (size_t)DFOUT * DFHID * 2;
  bf16_t* Wb2t = (bf16_t*)(ws + off); off += (size_t)DFOUT * DFHID * 2;
  bf16_t* Wr2t = (bf16_t*)(ws + off); off += (size_t)DFOUT * DFHID * 2;
  bf16_t* Wt1t = (bf16_t*)(ws + off); off += (size_t)DTHID * DTIN * 2;
  bf16_t* Wt2t = (bf16_t*)(ws + off); off += (size_t)DTOUT * DTHID * 2;

  // zero aggregate (graph-capturable)
  hipMemsetAsync(agg, 0, (size_t)NNODES * DTIN * 4, stream);

  // one-shot conversions
  cvt_x_kernel<<<2048, 256, 0, stream>>>(x, xbf, NNODES * DNODE / 4);
  cvt_wt_kernel<<<512, 256, 0, stream>>>(Wf1, Wf1t, DFIN,  DFHID, 0);
  cvt_wt_kernel<<<512, 256, 0, stream>>>(Wb1, Wb1t, DFIN,  DFHID, 1);  // swapped blocks
  cvt_wt_kernel<<<512, 256, 0, stream>>>(Wr1, Wr1t, DFIN,  DFHID, 0);
  cvt_wt_kernel<<<256, 256, 0, stream>>>(Wf2, Wf2t, DFHID, DFOUT, 0);
  cvt_wt_kernel<<<256, 256, 0, stream>>>(Wb2, Wb2t, DFHID, DFOUT, 0);
  cvt_wt_kernel<<<256, 256, 0, stream>>>(Wr2, Wr2t, DFHID, DFOUT, 0);
  cvt_wt_kernel<<<512, 256, 0, stream>>>(Wt1, Wt1t, DTIN,  DTHID, 0);
  cvt_wt_kernel<<<512, 256, 0, stream>>>(Wt2, Wt2t, DTHID, DTOUT, 0);

  // temporal edges: stage [x_future, x_past, attr]; fwd -> scatter(future, col 0),
  // bwd (row-swapped W1) -> scatter(past, col 256)
  edge_mlp_kernel<true><<<NEDGES / 64, 256, 0, stream>>>(
      xbf, ei + NEDGES /*future*/, ei /*past*/, eatt,
      Wf1t, Wb1t, bf1, bb1, Wf2t, Wb2t, bf2, bb2,
      agg, NEDGES, 0, 256);

  // frame edges: stage [x_early, x_later, attr]; one model scattered to both
  // endpoints at col 128
  edge_mlp_kernel<false><<<(NFEDGES + 63) / 64, 256, 0, stream>>>(
      xbf, fei /*early*/, fei + NFEDGES /*later*/, fatt,
      Wr1t, nullptr, br1, nullptr, Wr2t, nullptr, br2, nullptr,
      agg, NFEDGES, 128, 128);

  // final node MLP
  total_mlp_kernel<<<(NNODES + 63) / 64, 256, 0, stream>>>(
      agg, Wt1t, bt1, Wt2t, bt2, (float*)d_out, NNODES);
}
